// MultiHeadAttention_28089086116274
// MI455X (gfx1250) — matrix operand order repaired and checked by reading, not compiled, not run on hardware
//
#include <hip/hip_runtime.h>
#include <hip/hip_bf16.h>

// ---------------------------------------------------------------------------
// MI455X (gfx1250) fused multi-head attention, bf16 WMMA (f32 accumulate).
// Compute-bound (~206 GFLOP vs ~100MB HBM) -> WMMA pipe + async LDS staging.
// ---------------------------------------------------------------------------

typedef __attribute__((ext_vector_type(16))) __bf16 v16bf;
typedef __attribute__((ext_vector_type(8)))  float  v8f;
typedef __attribute__((__vector_size__(16))) int    v4i;

#define D_MODEL 1024
#define SEQ     4096
#define BATCH   2
#define NHEAD   16
#define HDIM    64
#define GSTR    72          // LDS row stride (bf16 elems): 144B rows -> bank spread

#define WMMA_BF16(a, b, c) \
  __builtin_amdgcn_wmma_f32_16x16x32_bf16(false, (a), false, (b), (short)0, (c), false, false)

__device__ __forceinline__ unsigned short f2bf(float f) {
  union { float f; unsigned int u; } v; v.f = f;
  unsigned int r = v.u + 0x7FFFu + ((v.u >> 16) & 1u);   // round-to-nearest-even
  return (unsigned short)(r >> 16);
}

// --- async global->LDS (CDNA5 GLOBAL_LOAD_ASYNC_TO_LDS_B128, ASYNCcnt) -----
__device__ __forceinline__ void async_cp16(const unsigned short* g, unsigned short* l) {
#if __has_builtin(__builtin_amdgcn_global_load_async_to_lds_b128)
  __builtin_amdgcn_global_load_async_to_lds_b128(
      (__attribute__((address_space(1))) v4i*)g,
      (__attribute__((address_space(3))) v4i*)l, 0, 0);
#else
  *reinterpret_cast<uint4*>(l) = *reinterpret_cast<const uint4*>(g);
#endif
}

__device__ __forceinline__ void wait_async() {
#if __has_builtin(__builtin_amdgcn_s_wait_asynccnt)
  __builtin_amdgcn_s_wait_asynccnt(0);
#else
  asm volatile("s_wait_asynccnt 0x0" ::: "memory");
#endif
}

// copy 32 bf16 (64B) = 4 async b128 transfers
__device__ __forceinline__ void cp_row32(const unsigned short* __restrict__ g,
                                         unsigned short* l) {
  async_cp16(g,      l);
  async_cp16(g + 8,  l + 8);
  async_cp16(g + 16, l + 16);
  async_cp16(g + 24, l + 24);
}

// A fragment (16x32 MxK, bf16): lane<16 -> M=lane, K=kk..kk+15;
// lane>=16 -> M=lane-16, K=kk+16..kk+31. Two 16B LDS reads.
__device__ __forceinline__ v16bf load_frag_a(const unsigned short* lds,
                                             int row_base, int kk, int stride) {
  const int lane = threadIdx.x & 31;
  const int hi = lane >> 4;
  const unsigned short* p = lds + (row_base + (lane & 15)) * stride;
  union { v16bf v; uint4 u[2]; } f;
  f.u[0] = *reinterpret_cast<const uint4*>(p + kk + hi * 16);
  f.u[1] = *reinterpret_cast<const uint4*>(p + kk + hi * 16 + 8);
  return f.v;
}

// B fragment (32x16 KxN, bf16): lane<16 -> N=lane, K=kk..kk+15;
// lane>=16 -> N=lane-16, K=kk+16..kk+31. 32 contiguous bytes per lane.
__device__ __forceinline__ v16bf load_frag_b(const unsigned short* lds,
                                             int n_base, int kk, int stride) {
  const int lane = threadIdx.x & 31;
  const int hi = lane >> 4;
  const unsigned short* p = lds + (n_base + (lane & 15)) * stride + kk + hi * 16;
  union { v16bf v; uint4 u[2]; } f;
  f.u[0] = *reinterpret_cast<const uint4*>(p);
  f.u[1] = *reinterpret_cast<const uint4*>(p + 8);
  return f.v;
}

__global__ void cvt_f32_bf16(const float* __restrict__ src,
                             unsigned short* __restrict__ dst, int n) {
  int i = blockIdx.x * blockDim.x + threadIdx.x;
  int stride = gridDim.x * blockDim.x;
  for (; i < n; i += stride) dst[i] = f2bf(src[i]);
}

// ---------------------------------------------------------------------------
// Y = X @ W^T.  X:[M][K] bf16, W:[N][K] bf16 (torch layout -> B operand is a
// contiguous W row along k).  Tile 128(M) x 64(N) x 64(K), 4 waves; each wave
// owns a 32x64 strip (2x4 v8f accumulators) so every B fragment feeds two
// WMMAs: 16 WMMAs per K-step at ~1.5 LDS b128 loads per WMMA.
// Double-buffered async LDS staging.
// mode 0: write bf16 to outb.  mode 1: write f32 + bias to outf.
// ---------------------------------------------------------------------------
__global__ void __launch_bounds__(128)
gemm_xwT(const unsigned short* __restrict__ X,
         const unsigned short* __restrict__ W,
         unsigned short* __restrict__ outb,
         float* __restrict__ outf,
         const float* __restrict__ bias,
         int Mdim, int Ndim, int Kdim, int mode) {
  __shared__ __align__(16) unsigned short Xs[2][128 * GSTR];
  __shared__ __align__(16) unsigned short Ws[2][64 * GSTR];

  const int tid  = threadIdx.x;
  const int lane = tid & 31;
  const int wave = tid >> 5;
  const int hi   = lane >> 4;
  const int nl   = lane & 15;
  const int m0   = blockIdx.y * 128;
  const int n0   = blockIdx.x * 64;

  v8f acc[2][4] = {};

  auto stage = [&](int k0, int buf) {
    // X tile: 128 rows x 64 k; one row per thread (8 async b128)
    const unsigned short* gx = X + (size_t)(m0 + tid) * Kdim + k0;
    unsigned short* lx = &Xs[buf][tid * GSTR];
    cp_row32(gx, lx);
    cp_row32(gx + 32, lx + 32);
    // W tile: 64 rows x 64 k; half row per thread
    const int wrow = tid >> 1, wcol = (tid & 1) * 32;
    cp_row32(W + (size_t)(n0 + wrow) * Kdim + k0 + wcol, &Ws[buf][wrow * GSTR + wcol]);
  };

  stage(0, 0);

  for (int k0 = 0; k0 < Kdim; k0 += 64) {
    const int buf = (k0 >> 6) & 1;
    wait_async();
    __syncthreads();                 // staged tile visible; prev reads done
    if (k0 + 64 < Kdim) stage(k0 + 64, buf ^ 1);

    const unsigned short* xs = Xs[buf];
    const unsigned short* ws = Ws[buf];
    v16bf aL0 = load_frag_a(xs, wave * 32,       0, GSTR);
    v16bf aL1 = load_frag_a(xs, wave * 32 + 16,  0, GSTR);
    v16bf aH0 = load_frag_a(xs, wave * 32,      32, GSTR);
    v16bf aH1 = load_frag_a(xs, wave * 32 + 16, 32, GSTR);
    {
      v16bf b0 = load_frag_b(ws,  0, 0, GSTR);
      v16bf b1 = load_frag_b(ws, 16, 0, GSTR);
      v16bf b2 = load_frag_b(ws, 32, 0, GSTR);
      v16bf b3 = load_frag_b(ws, 48, 0, GSTR);
      acc[0][0] = WMMA_BF16(aL0, b0, acc[0][0]);
      acc[1][0] = WMMA_BF16(aL1, b0, acc[1][0]);
      acc[0][1] = WMMA_BF16(aL0, b1, acc[0][1]);
      acc[1][1] = WMMA_BF16(aL1, b1, acc[1][1]);
      acc[0][2] = WMMA_BF16(aL0, b2, acc[0][2]);
      acc[1][2] = WMMA_BF16(aL1, b2, acc[1][2]);
      acc[0][3] = WMMA_BF16(aL0, b3, acc[0][3]);
      acc[1][3] = WMMA_BF16(aL1, b3, acc[1][3]);
    }
    {
      v16bf b0 = load_frag_b(ws,  0, 32, GSTR);
      v16bf b1 = load_frag_b(ws, 16, 32, GSTR);
      v16bf b2 = load_frag_b(ws, 32, 32, GSTR);
      v16bf b3 = load_frag_b(ws, 48, 32, GSTR);
      acc[0][0] = WMMA_BF16(aH0, b0, acc[0][0]);
      acc[1][0] = WMMA_BF16(aH1, b0, acc[1][0]);
      acc[0][1] = WMMA_BF16(aH0, b1, acc[0][1]);
      acc[1][1] = WMMA_BF16(aH1, b1, acc[1][1]);
      acc[0][2] = WMMA_BF16(aH0, b2, acc[0][2]);
      acc[1][2] = WMMA_BF16(aH1, b2, acc[1][2]);
      acc[0][3] = WMMA_BF16(aH0, b3, acc[0][3]);
      acc[1][3] = WMMA_BF16(aH1, b3, acc[1][3]);
    }
  }

  // C/D layout: VGPR r holds (M = r + 8*hi, N = lane%16)
#pragma unroll
  for (int s = 0; s < 2; ++s) {
#pragma unroll
    for (int r = 0; r < 8; ++r) {
      const int row = m0 + wave * 32 + s * 16 + r + hi * 8;
      const int c0 = n0 + nl, c1 = n0 + 16 + nl, c2 = n0 + 32 + nl, c3 = n0 + 48 + nl;
      if (mode == 0) {
        unsigned short* o = outb + (size_t)row * Ndim;
        o[c0] = f2bf(acc[s][0][r]);
        o[c1] = f2bf(acc[s][1][r]);
        o[c2] = f2bf(acc[s][2][r]);
        o[c3] = f2bf(acc[s][3][r]);
      } else {
        float* o = outf + (size_t)row * Ndim;
        o[c0] = acc[s][0][r] + bias[c0];
        o[c1] = acc[s][1][r] + bias[c1];
        o[c2] = acc[s][2][r] + bias[c2];
        o[c3] = acc[s][3][r] + bias[c3];
      }
    }
  }
}

// ---------------------------------------------------------------------------
// Fused causal flash attention for one (batch, head, 64-query tile).
// 4 waves; each wave owns a 16-row strip of the 64x64 O accumulator (regs).
// V is staged transposed in LDS so the P@V B-operand is contiguous per lane.
// Softmax: 2 threads per row (32 cols each) with LDS pair-combine.
// ---------------------------------------------------------------------------
__global__ void __launch_bounds__(128)
attn_fused(const unsigned short* __restrict__ Qb,
           const unsigned short* __restrict__ Kb,
           const unsigned short* __restrict__ Vb,
           unsigned short* __restrict__ Ctx) {
  __shared__ __align__(16) unsigned short Qs[64 * GSTR];
  __shared__ __align__(16) unsigned short Ks[64 * GSTR];
  __shared__ __align__(16) unsigned short VTs[64 * GSTR];
  __shared__ __align__(16) unsigned short Ps[64 * GSTR];
  __shared__ float Ss[64 * 65];
  __shared__ float red[128];
  __shared__ float m_s[64], l_s[64], c_s[64];

  const int tid  = threadIdx.x;
  const int lane = tid & 31;
  const int wave = tid >> 5;
  const int hi   = lane >> 4;
  const int nl   = lane & 15;

  const int qt = blockIdx.x;                 // query tile index (causal limit)
  const int h  = blockIdx.y;
  const int bi = blockIdx.z;
  const int q0 = qt * 64;
  const size_t base = ((size_t)bi * SEQ) * D_MODEL + (size_t)h * HDIM;

  const int lrow = tid >> 1;
  const int lcol = (tid & 1) * 32;

  { // load Q tile (64 queries x 64 head-dims), once
    const uint4* g = reinterpret_cast<const uint4*>(Qb + base + (size_t)(q0 + lrow) * D_MODEL + lcol);
    uint4* l = reinterpret_cast<uint4*>(Qs + lrow * GSTR + lcol);
    l[0] = g[0]; l[1] = g[1]; l[2] = g[2]; l[3] = g[3];
  }
  if (tid < 64) { m_s[tid] = -INFINITY; l_s[tid] = 0.f; }

  v8f o0 = {}, o1 = {}, o2 = {}, o3 = {};

  for (int j = 0; j <= qt; ++j) {
    __syncthreads();   // protect K/VT (still read by prev P@V) before reload
    // K tile, row-major (key x hd): async DMA straight into LDS
    cp_row32(Kb + base + (size_t)(j * 64 + lrow) * D_MODEL + lcol,
             Ks + lrow * GSTR + lcol);
    { // V tile transposed into LDS: VT[hd][key]
      union { uint4 u[4]; unsigned short s[32]; } vv;
      const uint4* g = reinterpret_cast<const uint4*>(Vb + base + (size_t)(j * 64 + lrow) * D_MODEL + lcol);
      vv.u[0] = g[0]; vv.u[1] = g[1]; vv.u[2] = g[2]; vv.u[3] = g[3];
#pragma unroll
      for (int i = 0; i < 32; ++i) VTs[(lcol + i) * GSTR + lrow] = vv.s[i];
    }
    wait_async();
    __syncthreads();

    { // S = (Q K^T) * 1/sqrt(64); wave computes rows wave*16..+15
      v8f s0 = {}, s1 = {}, s2 = {}, s3 = {};
      v16bf aL  = load_frag_a(Qs, wave * 16,  0, GSTR);
      v16bf aH  = load_frag_a(Qs, wave * 16, 32, GSTR);
      v16bf b0L = load_frag_b(Ks,  0,  0, GSTR);
      v16bf b1L = load_frag_b(Ks, 16,  0, GSTR);
      v16bf b2L = load_frag_b(Ks, 32,  0, GSTR);
      v16bf b3L = load_frag_b(Ks, 48,  0, GSTR);
      v16bf b0H = load_frag_b(Ks,  0, 32, GSTR);
      v16bf b1H = load_frag_b(Ks, 16, 32, GSTR);
      v16bf b2H = load_frag_b(Ks, 32, 32, GSTR);
      v16bf b3H = load_frag_b(Ks, 48, 32, GSTR);
      s0 = WMMA_BF16(aL, b0L, s0);
      s1 = WMMA_BF16(aL, b1L, s1);
      s2 = WMMA_BF16(aL, b2L, s2);
      s3 = WMMA_BF16(aL, b3L, s3);
      s0 = WMMA_BF16(aH, b0H, s0);
      s1 = WMMA_BF16(aH, b1H, s1);
      s2 = WMMA_BF16(aH, b2H, s2);
      s3 = WMMA_BF16(aH, b3H, s3);
#pragma unroll
      for (int r = 0; r < 8; ++r) {
        const int row = wave * 16 + r + hi * 8;
        Ss[row * 65 +  0 + nl] = s0[r] * 0.125f;
        Ss[row * 65 + 16 + nl] = s1[r] * 0.125f;
        Ss[row * 65 + 32 + nl] = s2[r] * 0.125f;
        Ss[row * 65 + 48 + nl] = s3[r] * 0.125f;
      }
    }
    __syncthreads();

    { // online softmax: 2 threads per row, 32 columns each
      const int r    = tid >> 1;
      const int half = tid & 1;
      const int cbase = half * 32;
      const int climit = (j == qt) ? r : 63;   // causal: key <= query
      const float* Srow = Ss + r * 65;
      const float mo = m_s[r];

      float pmx = -INFINITY;
      for (int c = 0; c < 32; ++c) {
        const int col = cbase + c;
        if (col <= climit) pmx = fmaxf(pmx, Srow[col]);
      }
      red[tid] = pmx;
      __syncthreads();
      const float mx = fmaxf(fmaxf(red[r * 2], red[r * 2 + 1]), mo);
      const float corr = __expf(mo - mx);
      float sum = 0.f;
      for (int c = 0; c < 32; ++c) {
        const int col = cbase + c;
        float pv = (col <= climit) ? __expf(Srow[col] - mx) : 0.f;
        Ps[r * GSTR + col] = f2bf(pv);
        sum += pv;
      }
      __syncthreads();          // all pair-reads of red done before rewrite
      red[tid] = sum;
      __syncthreads();
      if (half == 0) {
        m_s[r] = mx;
        l_s[r] = l_s[r] * corr + red[r * 2] + red[r * 2 + 1];
        c_s[r] = corr;
      }
    }
    __syncthreads();

    { // rescale O fragments by corr, then O += P @ V
#pragma unroll
      for (int r = 0; r < 8; ++r) {
        const float cr = c_s[wave * 16 + r + hi * 8];
        o0[r] *= cr; o1[r] *= cr; o2[r] *= cr; o3[r] *= cr;
      }
      v16bf aL  = load_frag_a(Ps, wave * 16,  0, GSTR);
      v16bf aH  = load_frag_a(Ps, wave * 16, 32, GSTR);
      v16bf b0L = load_frag_b(VTs,  0,  0, GSTR);
      v16bf b1L = load_frag_b(VTs, 16,  0, GSTR);
      v16bf b2L = load_frag_b(VTs, 32,  0, GSTR);
      v16bf b3L = load_frag_b(VTs, 48,  0, GSTR);
      v16bf b0H = load_frag_b(VTs,  0, 32, GSTR);
      v16bf b1H = load_frag_b(VTs, 16, 32, GSTR);
      v16bf b2H = load_frag_b(VTs, 32, 32, GSTR);
      v16bf b3H = load_frag_b(VTs, 48, 32, GSTR);
      o0 = WMMA_BF16(aL, b0L, o0);
      o1 = WMMA_BF16(aL, b1L, o1);
      o2 = WMMA_BF16(aL, b2L, o2);
      o3 = WMMA_BF16(aL, b3L, o3);
      o0 = WMMA_BF16(aH, b0H, o0);
      o1 = WMMA_BF16(aH, b1H, o1);
      o2 = WMMA_BF16(aH, b2H, o2);
      o3 = WMMA_BF16(aH, b3H, o3);
    }
  }

  // finalize: O /= l, write ctx (bf16) at [b, q0+row, h*64 + col]
#pragma unroll
  for (int r = 0; r < 8; ++r) {
    const int row = wave * 16 + r + hi * 8;
    const float inv = 1.0f / l_s[row];
    unsigned short* o = Ctx + base + (size_t)(q0 + row) * D_MODEL;
    o[ 0 + nl] = f2bf(o0[r] * inv);
    o[16 + nl] = f2bf(o1[r] * inv);
    o[32 + nl] = f2bf(o2[r] * inv);
    o[48 + nl] = f2bf(o3[r] * inv);
  }
}

// ---------------------------------------------------------------------------
extern "C" void kernel_launch(void* const* d_in, const int* in_sizes, int n_in,
                              void* d_out, int out_size, void* d_ws, size_t ws_size,
                              hipStream_t stream) {
  const float* x  = (const float*)d_in[0];
  const float* Wq = (const float*)d_in[1];
  const float* Wk = (const float*)d_in[2];
  const float* Wv = (const float*)d_in[3];
  const float* Wo = (const float*)d_in[4];
  const float* bo = (const float*)d_in[5];
  float* out = (float*)d_out;

  const int NX = BATCH * SEQ * D_MODEL;   // 8,388,608
  const int NW = D_MODEL * D_MODEL;       // 1,048,576

  char* p = (char*)d_ws;
  unsigned short* xb   = (unsigned short*)p; p += (size_t)NX * 2;
  unsigned short* wqb  = (unsigned short*)p; p += (size_t)NW * 2;
  unsigned short* wkb  = (unsigned short*)p; p += (size_t)NW * 2;
  unsigned short* wvb  = (unsigned short*)p; p += (size_t)NW * 2;
  unsigned short* wob  = (unsigned short*)p; p += (size_t)NW * 2;
  unsigned short* qb   = (unsigned short*)p; p += (size_t)NX * 2;
  unsigned short* kb   = (unsigned short*)p; p += (size_t)NX * 2;
  unsigned short* vb   = (unsigned short*)p; p += (size_t)NX * 2;
  unsigned short* ctxb = (unsigned short*)p; p += (size_t)NX * 2;

  cvt_f32_bf16<<<dim3((NX + 255) / 256), dim3(256), 0, stream>>>(x,  xb,  NX);
  cvt_f32_bf16<<<dim3((NW + 255) / 256), dim3(256), 0, stream>>>(Wq, wqb, NW);
  cvt_f32_bf16<<<dim3((NW + 255) / 256), dim3(256), 0, stream>>>(Wk, wkb, NW);
  cvt_f32_bf16<<<dim3((NW + 255) / 256), dim3(256), 0, stream>>>(Wv, wvb, NW);
  cvt_f32_bf16<<<dim3((NW + 255) / 256), dim3(256), 0, stream>>>(Wo, wob, NW);

  const dim3 ggrid(D_MODEL / 64, (BATCH * SEQ) / 128);  // 16 x 64
  gemm_xwT<<<ggrid, dim3(128), 0, stream>>>(xb, wqb, qb, nullptr, nullptr,
                                            BATCH * SEQ, D_MODEL, D_MODEL, 0);
  gemm_xwT<<<ggrid, dim3(128), 0, stream>>>(xb, wkb, kb, nullptr, nullptr,
                                            BATCH * SEQ, D_MODEL, D_MODEL, 0);
  gemm_xwT<<<ggrid, dim3(128), 0, stream>>>(xb, wvb, vb, nullptr, nullptr,
                                            BATCH * SEQ, D_MODEL, D_MODEL, 0);

  const dim3 agrid(SEQ / 64, NHEAD, BATCH);             // 64 x 16 x 2
  attn_fused<<<agrid, dim3(128), 0, stream>>>(qb, kb, vb, ctxb);

  gemm_xwT<<<ggrid, dim3(128), 0, stream>>>(ctxb, wob, nullptr, out, bo,
                                            BATCH * SEQ, D_MODEL, D_MODEL, 1);
}
